// ContinuousSpatialSSM_49898930045046
// MI455X (gfx1250) — matrix operand order repaired and checked by reading, not compiled, not run on hardware
//
#include <hip/hip_runtime.h>
#include <hip/hip_bf16.h>
#include <math.h>

// ---------------------------------------------------------------------------
// ContinuousSpatialSSM, K=3 closed-form unroll.
//   B=2, N=4096 (64x64), D=384, S=16.
// Phase 1: WMMA bf16 GEMM  x(8192x384) @ [W_ds|W_dd|W_B|W_C](384x800),
//          2x2 register-blocked (4 accumulators / wave), fused softplus
//          epilogue producing eps, c, Bmat, Cmat.
// Phase 2: three stencil kernels implementing the unrolled recurrence; all
//          S-contractions reduced to 16-term moments computed from LDS-cached
//          B/C rows; h (B,N,D,S) never materialized.
// ---------------------------------------------------------------------------

typedef __attribute__((ext_vector_type(16))) __bf16 v16bf;
typedef __attribute__((ext_vector_type(8)))  float  v8f;

#define DI   384
#define DS   16
#define NPIX 4096
#define ROWS 8192          // B*N
#define NCOL 800           // 384 + 384 + 16 + 16
#define KTIL 12            // 384/32
#define MTIL 512           // 8192/16
#define NTIL 50            // 800/16

// ---------------- pack x into WMMA A-fragment layout (16x32 bf16) ----------
// lane L (row = L&15), element i: k = (L>>4)*8 + (i&7) + (i>>3)*16
__global__ void k_pack_x(const float* __restrict__ x, __bf16* __restrict__ xp) {
    int t = blockIdx.x * blockDim.x + threadIdx.x;           // < MTIL*KTIL*32*16
    int i    = t & 15;
    int lane = (t >> 4) & 31;
    int ks   = (t >> 9) % KTIL;
    int mt   = t / (16 * 32 * KTIL);
    int row  = mt * 16 + (lane & 15);
    int k    = ks * 32 + ((lane >> 4) << 3) + (i & 7) + ((i >> 3) << 4);
    xp[t] = (__bf16)x[row * DI + k];
}

// ---------------- pack composite weight into B-fragment layout (32x16) -----
// lane L (col = L&15), element i: k = (L>>4)*8 + (i&7) + (i>>3)*16
__global__ void k_pack_w(const float* __restrict__ Wds, const float* __restrict__ Wdd,
                         const float* __restrict__ WB,  const float* __restrict__ WC,
                         __bf16* __restrict__ wp) {
    int t = blockIdx.x * blockDim.x + threadIdx.x;           // < NTIL*KTIL*32*16
    if (t >= NTIL * KTIL * 32 * 16) return;
    int i    = t & 15;
    int lane = (t >> 4) & 31;
    int ks   = (t >> 9) % KTIL;
    int nt   = t / (16 * 32 * KTIL);
    int k    = ks * 32 + ((lane >> 4) << 3) + (i & 7) + ((i >> 3) << 4);
    int col  = nt * 16 + (lane & 15);
    float w;
    if      (col < 384) w = Wds[k * DI + col];
    else if (col < 768) w = Wdd[k * DI + (col - 384)];
    else if (col < 784) w = WB [k * DS + (col - 768)];
    else                w = WC [k * DS + (col - 784)];
    wp[t] = (__bf16)w;
}

// ---------------- per-d prep: A = -softplus(A_log) (transposed), SA1, Dphys -
__global__ void k_prep(const float* __restrict__ A_log, const float* __restrict__ diff_raw,
                       float* __restrict__ At, float* __restrict__ SA1,
                       float* __restrict__ Dphys) {
    int d = blockIdx.x * blockDim.x + threadIdx.x;
    if (d >= DI) return;
    float sa = 0.f;
    for (int s = 0; s < DS; ++s) {
        float al = A_log[d * DS + s];
        float a  = -((al > 20.f) ? al : log1pf(__expf(al)));
        At[s * DI + d] = a;
        sa += a;
    }
    SA1[d]   = sa;
    Dphys[d] = 0.5f * (1.f / (1.f + __expf(-diff_raw[d])));
}

// ---------------- WMMA GEMM (2x2 register blocked) with fused epilogue -----
__global__ void k_gemm(const __bf16* __restrict__ xp, const __bf16* __restrict__ wp,
                       const float* __restrict__ b_ds, const float* __restrict__ b_dd,
                       const int* __restrict__ Kp,
                       float* __restrict__ eps, float* __restrict__ cdd,
                       float* __restrict__ Bm,  float* __restrict__ Cm) {
    int wid  = (blockIdx.x * blockDim.x + threadIdx.x) >> 5;   // global wave id
    int lane = threadIdx.x & 31;
    if (wid >= (MTIL / 2) * (NTIL / 2)) return;
    int mt0 = (wid / (NTIL / 2)) * 2;          // first of 2 M tiles
    int nt0 = (wid % (NTIL / 2)) * 2;          // first of 2 N tiles

    v8f a00 = {}, a01 = {}, a10 = {}, a11 = {};
    const v16bf* apA = (const v16bf*)xp + (size_t)(mt0    ) * KTIL * 32 + lane;
    const v16bf* apB = (const v16bf*)xp + (size_t)(mt0 + 1) * KTIL * 32 + lane;
    const v16bf* bpA = (const v16bf*)wp + (size_t)(nt0    ) * KTIL * 32 + lane;
    const v16bf* bpB = (const v16bf*)wp + (size_t)(nt0 + 1) * KTIL * 32 + lane;
#pragma unroll
    for (int ks = 0; ks < KTIL; ++ks) {
        v16bf fa0 = apA[ks * 32];
        v16bf fa1 = apB[ks * 32];
        v16bf fb0 = bpA[ks * 32];
        v16bf fb1 = bpB[ks * 32];
        a00 = __builtin_amdgcn_wmma_f32_16x16x32_bf16(false, fa0, false, fb0, (short)0, a00, false, false);
        a01 = __builtin_amdgcn_wmma_f32_16x16x32_bf16(false, fa0, false, fb1, (short)0, a01, false, false);
        a10 = __builtin_amdgcn_wmma_f32_16x16x32_bf16(false, fa1, false, fb0, (short)0, a10, false, false);
        a11 = __builtin_amdgcn_wmma_f32_16x16x32_bf16(false, fa1, false, fb1, (short)0, a11, false, false);
    }

    float dt = 1.0f / (float)Kp[0];
#pragma unroll
    for (int ti = 0; ti < 2; ++ti) {
#pragma unroll
        for (int tj = 0; tj < 2; ++tj) {
            v8f acc = (ti == 0) ? (tj == 0 ? a00 : a01) : (tj == 0 ? a10 : a11);
            int rb  = (mt0 + ti) * 16 + ((lane >> 4) << 3);
            int col = (nt0 + tj) * 16 + (lane & 15);
#pragma unroll
            for (int j = 0; j < 8; ++j) {
                int   row = rb + j;
                float z   = acc[j];
                if (col < 384) {
                    float e  = z + b_ds[col];
                    float sp = (e > 20.f) ? e : log1pf(__expf(e));
                    eps[row * DI + col] = dt * fminf(sp, 0.15f);
                } else if (col < 768) {
                    int   cc = col - 384;
                    float e  = z + b_dd[cc];
                    float sp = (e > 20.f) ? e : log1pf(__expf(e));
                    cdd[row * DI + cc] = dt * fminf(sp, 0.15f);
                } else if (col < 784) {
                    Bm[row * DS + (col - 768)] = z;
                } else {
                    Cm[row * DS + (col - 784)] = z;
                }
            }
        }
    }
}

// ---------------- stencil helpers ------------------------------------------
__device__ __forceinline__ float lap5(const float* __restrict__ H, int b, int n, int d) {
    const float* Hb = H + (size_t)b * NPIX * DI;
    int r = n >> 6, c = n & 63;
    float ctr = Hb[n * DI + d];
    float up  = (r >  0) ? Hb[(n - 64) * DI + d] : 0.f;
    float dn  = (r < 63) ? Hb[(n + 64) * DI + d] : 0.f;
    float lf  = (c >  0) ? Hb[(n -  1) * DI + d] : 0.f;
    float rt  = (c < 63) ? Hb[(n +  1) * DI + d] : 0.f;
    return up + dn + lf + rt - 4.f * ctr;
}

// ---------------- H0 = x * Q0 ----------------------------------------------
__global__ void k_h0(const float* __restrict__ x, const float* __restrict__ Bm,
                     float* __restrict__ H0) {
    int bn = blockIdx.x, d = threadIdx.x;
    __shared__ float sB[DS];
    if (d < DS) sB[d] = Bm[bn * DS + d];
    __syncthreads();
    float q0 = 0.f;
#pragma unroll
    for (int s = 0; s < DS; ++s) q0 += sB[s];
    size_t idx = (size_t)bn * DI + d;
    H0[idx] = x[idx] * q0;
}

// ---------------- step 1: g0, H1, y += c*g0*(R0+2eR1+e^2R2) ----------------
__global__ void k_step1(const float* __restrict__ x, const float* __restrict__ eps,
                        const float* __restrict__ cdd,
                        const float* __restrict__ Bm, const float* __restrict__ Cm,
                        const float* __restrict__ At, const float* __restrict__ Dphys,
                        const float* __restrict__ H0,
                        float* __restrict__ H1, float* __restrict__ g0f,
                        float* __restrict__ out) {
    int bn = blockIdx.x, d = threadIdx.x;
    int b = bn >> 12, n = bn & (NPIX - 1);
    __shared__ float sB[DS], sC[DS];
    if (d < DS)                sB[d]      = Bm[bn * DS + d];
    else if (d < 2 * DS)       sC[d - DS] = Cm[bn * DS + (d - DS)];
    __syncthreads();

    size_t idx = (size_t)bn * DI + d;
    float e = eps[idx], c = cdd[idx];
    float g0 = Dphys[d] * lap5(H0, b, n, d);

    float q0 = 0.f, q1 = 0.f, r0 = 0.f, r1 = 0.f, r2 = 0.f;
#pragma unroll
    for (int s = 0; s < DS; ++s) {
        float a = At[s * DI + d], Bs = sB[s], Cs = sC[s];
        q0 += Bs; q1 += a * Bs;
        r0 += Cs; r1 += a * Cs; r2 += a * a * Cs;
    }
    float xv = x[idx];
    H1[idx]  = xv * ((1.f + e) * q0 + e * q1) + 16.f * c * g0;
    g0f[idx] = g0;
    out[idx] = c * g0 * (r0 + 2.f * e * r1 + e * e * r2);   // initializes d_out
}

// ---------------- step 2: g1, H2, y += c*g1*(R0+eR1) -----------------------
__global__ void k_step2(const float* __restrict__ x, const float* __restrict__ eps,
                        const float* __restrict__ cdd,
                        const float* __restrict__ Bm, const float* __restrict__ Cm,
                        const float* __restrict__ At, const float* __restrict__ Dphys,
                        const float* __restrict__ SA1,
                        const float* __restrict__ H1, const float* __restrict__ g0f,
                        float* __restrict__ H2, float* __restrict__ out) {
    int bn = blockIdx.x, d = threadIdx.x;
    int b = bn >> 12, n = bn & (NPIX - 1);
    __shared__ float sB[DS], sC[DS];
    if (d < DS)                sB[d]      = Bm[bn * DS + d];
    else if (d < 2 * DS)       sC[d - DS] = Cm[bn * DS + (d - DS)];
    __syncthreads();

    size_t idx = (size_t)bn * DI + d;
    float e = eps[idx], c = cdd[idx];
    float g1 = Dphys[d] * lap5(H1, b, n, d);

    float q0 = 0.f, q1 = 0.f, q2 = 0.f, r0 = 0.f, r1 = 0.f;
#pragma unroll
    for (int s = 0; s < DS; ++s) {
        float a = At[s * DI + d], Bs = sB[s], Cs = sC[s];
        q0 += Bs; q1 += a * Bs; q2 += a * a * Bs;
        r0 += Cs; r1 += a * Cs;
    }
    float xv = x[idx];
    H2[idx] = xv * ((1.f + 2.f * e) * q0 + (2.f * e + e * e) * q1 + e * e * q2)
            + c * g0f[idx] * (16.f + e * SA1[d]) + 16.f * c * g1;
    out[idx] += c * g1 * (r0 + e * r1);
}

// ---------------- step 3: g2, final y --------------------------------------
__global__ void k_step3(const float* __restrict__ x, const float* __restrict__ eps,
                        const float* __restrict__ cdd,
                        const float* __restrict__ Bm, const float* __restrict__ Cm,
                        const float* __restrict__ At, const float* __restrict__ Dphys,
                        const float* __restrict__ Dparam,
                        const float* __restrict__ H2,
                        float* __restrict__ out) {
    int bn = blockIdx.x, d = threadIdx.x;
    int b = bn >> 12, n = bn & (NPIX - 1);
    __shared__ float sB[DS], sC[DS];
    if (d < DS)                sB[d]      = Bm[bn * DS + d];
    else if (d < 2 * DS)       sC[d - DS] = Cm[bn * DS + (d - DS)];
    __syncthreads();

    size_t idx = (size_t)bn * DI + d;
    float e = eps[idx], c = cdd[idx];
    float g2 = Dphys[d] * lap5(H2, b, n, d);

    float p0 = 0.f, p1 = 0.f, p2 = 0.f, p3 = 0.f, r0 = 0.f;
#pragma unroll
    for (int s = 0; s < DS; ++s) {
        float a = At[s * DI + d], bc = sB[s] * sC[s];
        p0 += bc; p1 += a * bc; p2 += a * a * bc; p3 += a * a * a * bc;
        r0 += sC[s];
    }
    float xv = x[idx];
    float poly = (1.f + 3.f * e) * p0 + 3.f * e * (1.f + e) * p1
               + e * e * (3.f + e) * p2 + e * e * e * p3;
    out[idx] += c * g2 * r0 + xv * poly + xv * Dparam[d];
}

// ---------------------------------------------------------------------------
extern "C" void kernel_launch(void* const* d_in, const int* in_sizes, int n_in,
                              void* d_out, int out_size, void* d_ws, size_t ws_size,
                              hipStream_t stream) {
    const float* x        = (const float*)d_in[0];
    const float* W_ds     = (const float*)d_in[1];
    const float* b_ds     = (const float*)d_in[2];
    const float* W_dd     = (const float*)d_in[3];
    const float* b_dd     = (const float*)d_in[4];
    const float* W_B      = (const float*)d_in[5];
    const float* W_C      = (const float*)d_in[6];
    const float* D_param  = (const float*)d_in[7];
    const float* A_log    = (const float*)d_in[8];
    const float* diff_raw = (const float*)d_in[9];
    const int*   Kp       = (const int*)  d_in[10];
    float* out = (float*)d_out;

    // workspace carve-up (all 256B-aligned)
    char* p = (char*)d_ws;
    auto carve = [&](size_t bytes) { char* q = p; p += (bytes + 255) & ~(size_t)255; return q; };
    __bf16* xp    = (__bf16*)carve((size_t)ROWS * DI * 2);
    __bf16* wp    = (__bf16*)carve((size_t)NCOL * DI * 2);
    float*  eps   = (float*) carve((size_t)ROWS * DI * 4);
    float*  cdd   = (float*) carve((size_t)ROWS * DI * 4);
    float*  Bm    = (float*) carve((size_t)ROWS * DS * 4);
    float*  Cm    = (float*) carve((size_t)ROWS * DS * 4);
    float*  At    = (float*) carve((size_t)DS * DI * 4);
    float*  SA1   = (float*) carve((size_t)DI * 4);
    float*  Dph   = (float*) carve((size_t)DI * 4);
    float*  H0    = (float*) carve((size_t)ROWS * DI * 4);   // reused as H2
    float*  H1    = (float*) carve((size_t)ROWS * DI * 4);
    float*  g0f   = (float*) carve((size_t)ROWS * DI * 4);

    // phase 0: packing + per-d prep
    k_pack_x<<<(MTIL * KTIL * 32 * 16) / 256, 256, 0, stream>>>(x, xp);
    k_pack_w<<<(NTIL * KTIL * 32 * 16 + 255) / 256, 256, 0, stream>>>(W_ds, W_dd, W_B, W_C, wp);
    k_prep<<<(DI + 255) / 256, 256, 0, stream>>>(A_log, diff_raw, At, SA1, Dph);

    // phase 1: WMMA GEMM (2x2 blocked: 6400 waves, 8 waves/block)
    k_gemm<<<((MTIL / 2) * (NTIL / 2)) / 8, 256, 0, stream>>>(xp, wp, b_ds, b_dd, Kp, eps, cdd, Bm, Cm);

    // phase 2: unrolled 3-step recurrence (one block per grid point)
    k_h0   <<<ROWS, DI, 0, stream>>>(x, Bm, H0);
    k_step1<<<ROWS, DI, 0, stream>>>(x, eps, cdd, Bm, Cm, At, Dph, H0, H1, g0f, out);
    k_step2<<<ROWS, DI, 0, stream>>>(x, eps, cdd, Bm, Cm, At, Dph, SA1, H1, g0f, /*H2=*/H0, out);
    k_step3<<<ROWS, DI, 0, stream>>>(x, eps, cdd, Bm, Cm, At, Dph, D_param, /*H2=*/H0, out);
}
